// GNN_1_2_46024869544456
// MI455X (gfx1250) — matrix operand order repaired and checked by reading, not compiled, not run on hardware
//
#include <hip/hip_runtime.h>
#include <hip/hip_bf16.h>

typedef __attribute__((ext_vector_type(16))) _Float16 v16h;
typedef __attribute__((ext_vector_type(8)))  float    v8f;

#define N_NODES  51200
#define E_EDGES  819200
#define G_GRAPHS 256
#define N2_SETS  102400
#define E2_EDGES 409600
#define D_EMB    64
#define L_LAYERS 5
#define NI2_F    36
#define NAF_F    40
#define EPS_BN   1e-5f

// ---------------------------------------------------------------------------
// Weight pack: f32 [K x Nc] row-major -> f16 WMMA B fragments, zero-padded to
// Kpad = ceil32(K).  Layout: frag(ks, tn, lane)[16] at ((ks*tN+tn)*32+lane)*16,
// element i holds W[ks*32 + 8*(lane>>4) + i + (i>=8 ? 8 : 0)][tn*16 + (lane&15)].
// ---------------------------------------------------------------------------
__global__ __launch_bounds__(256) void pack_w(
    const float* __restrict__ W, int ldw, int K, int Nc,
    _Float16* __restrict__ out)
{
  int t = blockIdx.x * blockDim.x + threadIdx.x;
  int tN = Nc >> 4;
  int Kpad = (K + 31) & ~31;
  int total = (Kpad >> 5) * tN * 32;
  if (t >= total) return;
  int lane = t & 31;
  int tn = (t >> 5) % tN;
  int ks = (t >> 5) / tN;
  int half = lane >> 4;
  int n = (tn << 4) + (lane & 15);
  _Float16* o = out + (size_t)t * 16;
#pragma unroll
  for (int i = 0; i < 16; ++i) {
    int kb = (ks << 5) + half * 8 + i + ((i >= 8) ? 8 : 0);
    int kc = (kb < K) ? kb : 0;
    float v = W[kc * ldw + n];
    o[i] = (_Float16)((kb < K) ? v : 0.0f);
  }
}

// ---------------------------------------------------------------------------
// WMMA GEMM: C = act((acc?C:0) + A[MxK]@W + bias). One wave = one 16x16 tile.
// A: f32 row-major (vector loads, branch-free); B: pre-packed f16 fragments.
// GUARD=true handles K not multiple of 32 via clamped (in-bounds) loads.
// ---------------------------------------------------------------------------
template <bool GUARD>
__global__ __launch_bounds__(256) void gemm_wmma(
    const float* __restrict__ A, int lda,
    const _Float16* __restrict__ Bp,
    const float* __restrict__ bias,
    float* __restrict__ C, int ldc,
    int M, int Kreal, int Kpad, int Nc, int relu, int acc)
{
  int wid = (int)((blockIdx.x * blockDim.x + threadIdx.x) >> 5);
  int tN = Nc >> 4;
  int tM = M >> 4;
  if (wid >= tM * tN) return;
  int tn = wid % tN, tm = wid / tN;
  int lane = threadIdx.x & 31;
  int half = lane >> 4, r = lane & 15;
  int m = (tm << 4) + r;
  int n = (tn << 4) + r;

  const float* arow = A + (size_t)m * lda;
  const _Float16* bptr = Bp + ((size_t)tn * 32 + lane) * 16;
  const size_t bstride = (size_t)tN * 32 * 16;
  const int nsteps = Kpad >> 5;

  v8f cacc = {};
  for (int ks = 0; ks < nsteps; ++ks) {
    int k0 = ks << 5;
    v16h a;
    if (GUARD) {
#pragma unroll
      for (int i = 0; i < 16; ++i) {
        int ka = k0 + i + ((i >= 8) ? 8 : 0) + half * 8;
        int kc = (ka < Kreal) ? ka : 0;       // clamp -> load stays in bounds
        float v = arow[kc];                    // unconditional load
        a[i] = (_Float16)((ka < Kreal) ? v : 0.0f);
      }
    } else {
      const float* ap = arow + k0 + half * 8;
      float4 a0 = *(const float4*)(ap);
      float4 a1 = *(const float4*)(ap + 4);
      float4 a2 = *(const float4*)(ap + 16);
      float4 a3 = *(const float4*)(ap + 20);
      a[0]  = (_Float16)a0.x; a[1]  = (_Float16)a0.y;
      a[2]  = (_Float16)a0.z; a[3]  = (_Float16)a0.w;
      a[4]  = (_Float16)a1.x; a[5]  = (_Float16)a1.y;
      a[6]  = (_Float16)a1.z; a[7]  = (_Float16)a1.w;
      a[8]  = (_Float16)a2.x; a[9]  = (_Float16)a2.y;
      a[10] = (_Float16)a2.z; a[11] = (_Float16)a2.w;
      a[12] = (_Float16)a3.x; a[13] = (_Float16)a3.y;
      a[14] = (_Float16)a3.z; a[15] = (_Float16)a3.w;
    }
    v16h b = *(const v16h*)(bptr);             // 2x global_load_b128
    bptr += bstride;
    cacc = __builtin_amdgcn_wmma_f32_16x16x32_f16(
        false, a, false, b, (short)0, cacc, false, false);
  }
#pragma unroll
  for (int j = 0; j < 8; ++j) {
    int mm = (tm << 4) + j + half * 8;
    float v = cacc[j];
    if (bias) v += bias[n];
    if (acc)  v += C[(size_t)mm * ldc + n];
    if (relu) v = fmaxf(v, 0.f);
    C[(size_t)mm * ldc + n] = v;
  }
}

// ---------------------------------------------------------------------------
// GIN scatter: agg[dst] += h[src] + eemb1[a0] + eemb2[a1] (+ N self loops
// with attr (4,0)).  One thread = one (edge, 4-channel group).
// ---------------------------------------------------------------------------
__global__ __launch_bounds__(256) void gin_scatter(
    const float* __restrict__ h,
    const int* __restrict__ ei,    // [2, E]
    const int* __restrict__ ea,    // [E, 2]
    const float* __restrict__ ee1, // [6, D] (this layer)
    const float* __restrict__ ee2, // [3, D]
    float* __restrict__ agg)
{
  int t = blockIdx.x * blockDim.x + threadIdx.x;
  const int total = (E_EDGES + N_NODES) * 16;
  if (t >= total) return;
  int c = (t & 15) * 4;
  int e = t >> 4;
  int s, d, a0, a1;
  if (e < E_EDGES) {
    s = ei[e]; d = ei[E_EDGES + e];
    a0 = ea[e * 2]; a1 = ea[e * 2 + 1];
  } else {
    s = d = e - E_EDGES; a0 = 4; a1 = 0;
  }
  float4 hv = *(const float4*)(h + s * D_EMB + c);
  float4 e1 = *(const float4*)(ee1 + a0 * D_EMB + c);
  float4 e2 = *(const float4*)(ee2 + a1 * D_EMB + c);
  float* dst = agg + d * D_EMB + c;
  atomicAdd(dst + 0, hv.x + e1.x + e2.x);
  atomicAdd(dst + 1, hv.y + e1.y + e2.y);
  atomicAdd(dst + 2, hv.z + e1.z + e2.z);
  atomicAdd(dst + 3, hv.w + e1.w + e2.w);
}

// conv1 scatter over virtual concat z = [xp | iso]; agg2 has padded lda=128
__global__ __launch_bounds__(256) void conv1_scatter(
    const float* __restrict__ xp, const float* __restrict__ iso,
    const int* __restrict__ ei2, float* __restrict__ agg2)
{
  int t = blockIdx.x * blockDim.x + threadIdx.x;
  const int total = E2_EDGES * 25;             // 25 groups of 4 channels
  if (t >= total) return;
  int c = (t % 25) * 4;
  int e = t / 25;
  int s = ei2[e], d = ei2[E2_EDGES + e];
  float4 v = (c < D_EMB) ? *(const float4*)(xp + s * D_EMB + c)
                         : *(const float4*)(iso + s * NI2_F + (c - D_EMB));
  float* dst = agg2 + d * 128 + c;
  atomicAdd(dst + 0, v.x);
  atomicAdd(dst + 1, v.y);
  atomicAdd(dst + 2, v.z);
  atomicAdd(dst + 3, v.w);
}

// conv2 scatter, D=64
__global__ __launch_bounds__(256) void conv2_scatter(
    const float* __restrict__ z, const int* __restrict__ ei2,
    float* __restrict__ agg3)
{
  int t = blockIdx.x * blockDim.x + threadIdx.x;
  const int total = E2_EDGES * 16;
  if (t >= total) return;
  int c = (t & 15) * 4;
  int e = t >> 4;
  int s = ei2[e], d = ei2[E2_EDGES + e];
  float4 v = *(const float4*)(z + s * D_EMB + c);
  float* dst = agg3 + d * D_EMB + c;
  atomicAdd(dst + 0, v.x);
  atomicAdd(dst + 1, v.y);
  atomicAdd(dst + 2, v.z);
  atomicAdd(dst + 3, v.w);
}

// batchnorm statistics: block per channel -> {mu[64], rstd[64]}
__global__ __launch_bounds__(256) void bn_reduce(
    const float* __restrict__ x, float* __restrict__ stats)
{
  int c = blockIdx.x;
  float s = 0.f, sq = 0.f;
  for (int i = threadIdx.x; i < N_NODES; i += 256) {
    float v = x[(size_t)i * D_EMB + c];
    s += v; sq += v * v;
  }
  __shared__ float ss[256], s2[256];
  ss[threadIdx.x] = s; s2[threadIdx.x] = sq;
  __syncthreads();
  for (int o = 128; o > 0; o >>= 1) {
    if (threadIdx.x < o) {
      ss[threadIdx.x] += ss[threadIdx.x + o];
      s2[threadIdx.x] += s2[threadIdx.x + o];
    }
    __syncthreads();
  }
  if (threadIdx.x == 0) {
    float mu  = ss[0] / (float)N_NODES;
    float var = s2[0] / (float)N_NODES - mu * mu;
    stats[c]         = mu;
    stats[D_EMB + c] = rsqrtf(var + EPS_BN);
  }
}

__global__ __launch_bounds__(256) void bn_apply(
    const float* __restrict__ x, const float* __restrict__ stats,
    const float* __restrict__ gamma, const float* __restrict__ beta,
    float* __restrict__ out, int relu)
{
  int t = blockIdx.x * blockDim.x + threadIdx.x;
  const int total = N_NODES * 16;
  if (t >= total) return;
  int c = (t & 15) * 4;
  float4 v = *(const float4*)(x + (size_t)t * 4);
  float4 o;
  o.x = (v.x - stats[c + 0]) * stats[D_EMB + c + 0] * gamma[c + 0] + beta[c + 0];
  o.y = (v.y - stats[c + 1]) * stats[D_EMB + c + 1] * gamma[c + 1] + beta[c + 1];
  o.z = (v.z - stats[c + 2]) * stats[D_EMB + c + 2] * gamma[c + 2] + beta[c + 2];
  o.w = (v.w - stats[c + 3]) * stats[D_EMB + c + 3] * gamma[c + 3] + beta[c + 3];
  if (relu) {
    o.x = fmaxf(o.x, 0.f); o.y = fmaxf(o.y, 0.f);
    o.z = fmaxf(o.z, 0.f); o.w = fmaxf(o.w, 0.f);
  }
  *(float4*)(out + (size_t)t * 4) = o;
}

// avg_pool onto 2-sets: exactly 2 assignment members per set
__global__ __launch_bounds__(256) void pair_mean(
    const float* __restrict__ h, const int* __restrict__ a0,
    float* __restrict__ xp)
{
  int t = blockIdx.x * blockDim.x + threadIdx.x;
  const int total = N2_SETS * 16;
  if (t >= total) return;
  int c = (t & 15) * 4;
  int i = t >> 4;
  int u = a0[2 * i], v = a0[2 * i + 1];
  float4 x = *(const float4*)(h + u * D_EMB + c);
  float4 y = *(const float4*)(h + v * D_EMB + c);
  float4 o;
  o.x = 0.5f * (x.x + y.x); o.y = 0.5f * (x.y + y.y);
  o.z = 0.5f * (x.z + y.z); o.w = 0.5f * (x.w + y.w);
  *(float4*)(xp + i * D_EMB + c) = o;
}

// contiguous-segment mean pool (batch / batch_2 are repeat(arange(G)))
__global__ __launch_bounds__(256) void pool_mean(
    const float* __restrict__ in, float* __restrict__ out,
    int rows_per, int ldo)
{
  int t = blockIdx.x * blockDim.x + threadIdx.x;
  if (t >= G_GRAPHS * D_EMB) return;
  int c = t & (D_EMB - 1);
  int g = t >> 6;
  const float* p = in + (size_t)g * rows_per * D_EMB + c;
  float s = 0.f;
  for (int i = 0; i < rows_per; ++i) s += p[(size_t)i * D_EMB];
  out[g * ldo + c] = s / (float)rows_per;
}

// final 16 -> 1 projection
__global__ __launch_bounds__(256) void final_dot(
    const float* __restrict__ m3, const float* __restrict__ w,
    const float* __restrict__ b, float* __restrict__ out)
{
  int g = blockIdx.x * blockDim.x + threadIdx.x;
  if (g >= G_GRAPHS) return;
  float s = b[0];
#pragma unroll
  for (int j = 0; j < 16; ++j) s += m3[g * 16 + j] * w[j];
  out[g] = s;
}

// ---------------------------------------------------------------------------
static inline unsigned cdivu(long long a, int b) {
  return (unsigned)((a + b - 1) / b);
}
static inline int kpad32(int K) { return (K + 31) & ~31; }

static inline void launch_pack(const float* W, int ldw, int K, int Nc,
                               _Float16* out, hipStream_t s) {
  int total = (kpad32(K) >> 5) * (Nc >> 4) * 32;
  pack_w<<<cdivu(total, 256), 256, 0, s>>>(W, ldw, K, Nc, out);
}

static inline void launch_gemm(const float* A, int lda, const _Float16* Bp,
                               const float* bias, float* C, int ldc,
                               int M, int K, int Nc, int relu, int acc,
                               hipStream_t s) {
  long long tiles = (long long)(M >> 4) * (Nc >> 4);
  unsigned blocks = cdivu(tiles * 32, 256);
  int Kp = kpad32(K);
  if (K == Kp)
    gemm_wmma<false><<<blocks, 256, 0, s>>>(A, lda, Bp, bias, C, ldc, M, K, Kp, Nc, relu, acc);
  else
    gemm_wmma<true><<<blocks, 256, 0, s>>>(A, lda, Bp, bias, C, ldc, M, K, Kp, Nc, relu, acc);
}

extern "C" void kernel_launch(void* const* d_in, const int* in_sizes, int n_in,
                              void* d_out, int out_size, void* d_ws, size_t ws_size,
                              hipStream_t stream) {
  (void)in_sizes; (void)n_in; (void)out_size; (void)ws_size;
  // ---- inputs (setup_inputs dict order) ----
  const float* x        = (const float*)d_in[0];
  const int*   ei       = (const int*)d_in[1];
  const int*   ea       = (const int*)d_in[2];
  const float* iso      = (const float*)d_in[4];
  const int*   ei2      = (const int*)d_in[5];
  const int*   assign   = (const int*)d_in[6];
  const float* emb_W    = (const float*)d_in[8];
  const float* emb_b    = (const float*)d_in[9];
  const float* gin_W1   = (const float*)d_in[10];
  const float* gin_b1   = (const float*)d_in[11];
  const float* gin_W2   = (const float*)d_in[12];
  const float* gin_b2   = (const float*)d_in[13];
  const float* eemb1    = (const float*)d_in[14];
  const float* eemb2    = (const float*)d_in[15];
  const float* bn_gamma = (const float*)d_in[16];
  const float* bn_beta  = (const float*)d_in[17];
  const float* i1_relW  = (const float*)d_in[18];
  const float* i1_relb  = (const float*)d_in[19];
  const float* i1_rootW = (const float*)d_in[20];
  const float* i2_relW  = (const float*)d_in[21];
  const float* i2_relb  = (const float*)d_in[22];
  const float* i2_rootW = (const float*)d_in[23];
  const float* ro_W0    = (const float*)d_in[24];
  const float* ro_b0    = (const float*)d_in[25];
  const float* ro_W1    = (const float*)d_in[26];
  const float* ro_b1    = (const float*)d_in[27];
  const float* ro_W2    = (const float*)d_in[28];
  const float* ro_b2    = (const float*)d_in[29];
  const float* last_W   = (const float*)d_in[30];
  const float* last_b   = (const float*)d_in[31];
  float* out = (float*)d_out;

  // ---- workspace layout (~132 MB) ----
  char* ws = (char*)d_ws;
  const size_t SZ_H    = (size_t)N_NODES * 64 * 4;    // 13,107,200
  const size_t SZ_T1   = (size_t)N_NODES * 128 * 4;   // 26,214,400 (== N2*64*4)
  const size_t SZ_AGG2 = (size_t)N2_SETS * 128 * 4;   // 52,428,800 (padded lda=128)
  const size_t SZ_TC   = (size_t)N2_SETS * 64 * 4;    // 26,214,400
  float* h    = (float*)(ws);
  float* agg  = (float*)(ws + SZ_H);
  float* t1   = (float*)(ws + 2 * SZ_H);                       // GIN mid / xp / t2
  float* agg2 = (float*)(ws + 2 * SZ_H + SZ_T1);               // conv1/conv2 agg
  float* tc   = (float*)(ws + 2 * SZ_H + SZ_T1 + SZ_AGG2);     // conv1 out (zp2)
  char*  smal =        ws + 2 * SZ_H + SZ_T1 + SZ_AGG2 + SZ_TC;
  float* stats = (float*)(smal);                 // mu[64], rstd[64]
  float* mbuf  = (float*)(smal + 1024);          // [256,128]
  float* m1    = mbuf + 256 * 128;
  float* m2    = m1 + 256 * 64;
  float* m3    = m2 + 256 * 32;
  // packed f16 weight fragments
  _Float16* pk = (_Float16*)(m3 + 256 * 16);
  _Float16* p_emb = pk;                 pk += (size_t)kpad32(NAF_F) * 64;     // 64x64
  _Float16* p_w1[L_LAYERS];
  _Float16* p_w2[L_LAYERS];
  for (int l = 0; l < L_LAYERS; ++l) { p_w1[l] = pk; pk += 64 * 128; }
  for (int l = 0; l < L_LAYERS; ++l) { p_w2[l] = pk; pk += 128 * 64; }
  _Float16* p_i1rel  = pk;  pk += 128 * 64;                    // K=100 -> 128
  _Float16* p_i1rtA  = pk;  pk += 64 * 64;
  _Float16* p_i1rtB  = pk;  pk += (size_t)kpad32(NI2_F) * 64;  // K=36 -> 64
  _Float16* p_i2rel  = pk;  pk += 64 * 64;
  _Float16* p_i2rt   = pk;  pk += 64 * 64;
  _Float16* p_ro0    = pk;  pk += 128 * 64;
  _Float16* p_ro1    = pk;  pk += 64 * 32;
  _Float16* p_ro2    = pk;  pk += 32 * 16;

  // ---- pack all weights to f16 fragments (tiny) ----
  launch_pack(emb_W, 64, NAF_F, 64, p_emb, stream);
  for (int l = 0; l < L_LAYERS; ++l) {
    launch_pack(gin_W1 + l * 64 * 128, 128, 64, 128, p_w1[l], stream);
    launch_pack(gin_W2 + l * 128 * 64, 64, 128, 64, p_w2[l], stream);
  }
  launch_pack(i1_relW, 64, 100, 64, p_i1rel, stream);
  launch_pack(i1_rootW, 64, 64, 64, p_i1rtA, stream);
  launch_pack(i1_rootW + 64 * 64, 64, NI2_F, 64, p_i1rtB, stream);
  launch_pack(i2_relW, 64, 64, 64, p_i2rel, stream);
  launch_pack(i2_rootW, 64, 64, 64, p_i2rt, stream);
  launch_pack(ro_W0, 64, 128, 64, p_ro0, stream);
  launch_pack(ro_W1, 32, 64, 32, p_ro1, stream);
  launch_pack(ro_W2, 16, 32, 16, p_ro2, stream);

  // ---- embed: h = relu(x @ emb_W + emb_b), K=40 (guarded A path) ----
  launch_gemm(x, NAF_F, p_emb, emb_b, h, 64, N_NODES, NAF_F, 64, 1, 0, stream);

  // ---- 5 GIN layers ----
  const int etot16 = (E_EDGES + N_NODES) * 16;
  for (int l = 0; l < L_LAYERS; ++l) {
    hipMemsetAsync(agg, 0, SZ_H, stream);
    gin_scatter<<<cdivu(etot16, 256), 256, 0, stream>>>(
        h, ei, ea, eemb1 + l * 6 * 64, eemb2 + l * 3 * 64, agg);
    launch_gemm(agg, 64, p_w1[l], gin_b1 + l * 128, t1, 128,
                N_NODES, 64, 128, 1, 0, stream);
    launch_gemm(t1, 128, p_w2[l], gin_b2 + l * 64, agg, 64,
                N_NODES, 128, 64, 0, 0, stream);
    bn_reduce<<<64, 256, 0, stream>>>(agg, stats);
    bn_apply<<<cdivu(N_NODES * 16, 256), 256, 0, stream>>>(
        agg, stats, bn_gamma + l * 64, bn_beta + l * 64, h, (l < L_LAYERS - 1));
  }

  // ---- graph mean pool x1 -> mbuf[:, 0:64] ----
  pool_mean<<<cdivu(G_GRAPHS * 64, 256), 256, 0, stream>>>(h, mbuf, 200, 128);

  // ---- 2-set branch ----
  float* xp = t1;
  pair_mean<<<cdivu(N2_SETS * 16, 256), 256, 0, stream>>>(h, assign, xp);

  // conv1: tc = relu(agg2@Wrel + brel + xp@Wroot[0:64] + iso@Wroot[64:100])
  hipMemsetAsync(agg2, 0, SZ_AGG2, stream);   // zeros padding cols [100,128)
  conv1_scatter<<<cdivu(E2_EDGES * 25, 256), 256, 0, stream>>>(xp, iso, ei2, agg2);
  launch_gemm(agg2, 128, p_i1rel, i1_relb, tc, 64, N2_SETS, 128, 64, 0, 0, stream);
  launch_gemm(xp, 64, p_i1rtA, nullptr, tc, 64, N2_SETS, 64, 64, 0, 1, stream);
  launch_gemm(iso, NI2_F, p_i1rtB, nullptr, tc, 64, N2_SETS, NI2_F, 64, 1, 1, stream);

  // conv2: t2 = relu(agg3@Wrel + brel + tc@Wroot)
  float* agg3 = agg2;  // reuse, stride 64 now
  float* t2 = t1;      // xp dead
  hipMemsetAsync(agg3, 0, (size_t)N2_SETS * 64 * 4, stream);
  conv2_scatter<<<cdivu(E2_EDGES * 16, 256), 256, 0, stream>>>(tc, ei2, agg3);
  launch_gemm(agg3, 64, p_i2rel, i2_relb, t2, 64, N2_SETS, 64, 64, 0, 0, stream);
  launch_gemm(tc, 64, p_i2rt, nullptr, t2, 64, N2_SETS, 64, 64, 1, 1, stream);

  // x2 -> mbuf[:, 64:128]
  pool_mean<<<cdivu(G_GRAPHS * 64, 256), 256, 0, stream>>>(t2, mbuf + 64, 400, 128);

  // ---- readout MLP ----
  launch_gemm(mbuf, 128, p_ro0, ro_b0, m1, 64, G_GRAPHS, 128, 64, 1, 0, stream);
  launch_gemm(m1, 64, p_ro1, ro_b1, m2, 32, G_GRAPHS, 64, 32, 1, 0, stream);
  launch_gemm(m2, 32, p_ro2, ro_b2, m3, 16, G_GRAPHS, 32, 16, 1, 0, stream);
  final_dot<<<1, 256, 0, stream>>>(m3, last_W, last_b, out);
}
